// LSTMCell_18622978196287
// MI455X (gfx1250) — matrix operand order repaired and checked by reading, not compiled, not run on hardware
//
#include <hip/hip_runtime.h>
#include <hip/hip_bf16.h>

// LSTM: T=512, B=64, H=1024.  gates = [x_t, h] @ W^T + b, per-step GEMM
// [64,2048]x[2048,4096] on v_wmma_f32_16x16x32_bf16.  x and h are stored in
// fragment-native bf16 layout so the hot K-loop is pure b128-load + WMMA.

typedef __attribute__((ext_vector_type(16))) __bf16          v16bf;
typedef __attribute__((ext_vector_type(8)))  float           v8f;
typedef __attribute__((ext_vector_type(4)))  float           f4;
typedef __attribute__((ext_vector_type(8)))  unsigned short  us8;
typedef __attribute__((ext_vector_type(16))) unsigned short  us16;

#define T_STEPS 512
#define BATCH   64
#define HID     1024
#define K2H     2048
#define NSTRIP  64   // HID/16 column strips
#define NKT     64   // K2H/32 k-tiles
#define FRAG_H  (32 * 16)                 // halves per (kt,*) fragment slab: 32 lanes x 16
#define XSTEP   (4 * 32 * FRAG_H)         // packed halves per timestep: [mt][kt][lane][16]

__device__ __forceinline__ unsigned short f2bf(float f) {
  unsigned int u = __float_as_uint(f);
  u += 0x7FFFu + ((u >> 16) & 1u);           // round-to-nearest-even
  return (unsigned short)(u >> 16);
}
__device__ __forceinline__ float sig_(float x)  { return 1.f / (1.f + __expf(-x)); }
__device__ __forceinline__ float tanh_(float x) { return 1.f - 2.f / (1.f + __expf(2.f * x)); }

// ---------------------------------------------------------------------------
// Pack W into WMMA B-fragment layout: [gate][strip][kt][lane][16 halves],
// lane l -> column l&15; lanes<16 K-offsets {0..7,16..23}, lanes>=16 {8..15,24..31}.
// ---------------------------------------------------------------------------
__global__ void pack_w(const float* __restrict__ Wi, const float* __restrict__ Wf,
                       const float* __restrict__ Wc, const float* __restrict__ Wo,
                       unsigned short* __restrict__ Wb) {
  int tid = blockIdx.x * blockDim.x + threadIdx.x;   // 4*64*64*32 = 524288
  int l  = tid & 31;
  int kt = (tid >> 5) & 63;
  int s  = (tid >> 11) & 63;
  int g  = tid >> 17;
  const float* Wg = (g == 0) ? Wi : (g == 1) ? Wf : (g == 2) ? Wc : Wo;
  int n  = s * 16 + (l & 15);
  int k0 = kt * 32 + (l >> 4) * 8;
  const float* src = Wg + (size_t)n * K2H + k0;
  us8 a, b;
#pragma unroll
  for (int j = 0; j < 8; ++j) { a[j] = f2bf(src[j]); b[j] = f2bf(src[16 + j]); }
  us8* dst = (us8*)(Wb + (size_t)tid * 16);
  dst[0] = a; dst[1] = b;
}

// ---------------------------------------------------------------------------
// Pack x [T,B,H] fp32 into WMMA A-fragment layout: [t][mt][kt][lane][16],
// lane l -> row l&15; lanes<16 K-offsets {0..7,16..23}, lanes>=16 {8..15,24..31}.
// ---------------------------------------------------------------------------
__global__ void pack_x(const float* __restrict__ x, unsigned short* __restrict__ xb) {
  size_t tid = (size_t)blockIdx.x * blockDim.x + threadIdx.x;  // 512*4*32*32 = 2097152
  int l  = (int)(tid & 31);
  int kt = (int)((tid >> 5) & 31);
  int mt = (int)((tid >> 10) & 3);
  int t  = (int)(tid >> 12);
  int row = mt * 16 + (l & 15);
  int kh  = (l >> 4) * 8;
  const float* src = x + ((size_t)t * BATCH + row) * HID + kt * 32 + kh;
  us8 a, b;
#pragma unroll
  for (int j = 0; j < 8; ++j) { a[j] = f2bf(src[j]); b[j] = f2bf(src[16 + j]); }
  us8* dst = (us8*)(xb + tid * 16);
  dst[0] = a; dst[1] = b;
}

// zero h0 (packed bf16) and c0 (f32)
__global__ void init_state(unsigned short* __restrict__ h0, float* __restrict__ c0) {
  int i = blockIdx.x * blockDim.x + threadIdx.x;     // 65536
  h0[i] = 0; c0[i] = 0.f;
}

// ---------------------------------------------------------------------------
// One LSTM timestep (fast path: packed x).  64 blocks (column strips of 16)
// x 128 threads (4 wave32s, wave = batch M-tile).  Per k-step: one A fragment
// (32B load) shared by 4 gate WMMAs, each with a 32B B-fragment load.
// ---------------------------------------------------------------------------
__global__ void lstm_step(const unsigned short* __restrict__ xb,   // packed [t]...
                          const float* __restrict__ bi, const float* __restrict__ bfv,
                          const float* __restrict__ bc, const float* __restrict__ bo,
                          const unsigned short* __restrict__ Wb,
                          const unsigned short* __restrict__ h_prev,  // packed A layout
                          unsigned short* __restrict__ h_next,        // packed A layout
                          float* __restrict__ cst,
                          float* __restrict__ hseq,
                          float* __restrict__ hT, float* __restrict__ cT,
                          int t, int last) {
  const int lane  = threadIdx.x & 31;
  const int mt    = threadIdx.x >> 5;      // wave id = batch tile 0..3
  const int strip = blockIdx.x;            // 0..63
  const int idx16 = lane & 15;
  const int hsel  = lane >> 4;             // 0/1 half-wave
  const int ncol  = strip * 16 + idx16;    // output column this lane owns

  const unsigned short* xfr = xb + (size_t)t * XSTEP + ((size_t)mt * 32) * FRAG_H + (size_t)lane * 16;
  const unsigned short* hfr = h_prev      + ((size_t)mt * 32) * FRAG_H + (size_t)lane * 16;

  const size_t gstride = (size_t)NSTRIP * NKT * FRAG_H;         // per-gate packed stride
  const unsigned short* wbase = Wb + (size_t)strip * NKT * FRAG_H + (size_t)lane * 16;

  v8f acc0 = {}, acc1 = {}, acc2 = {}, acc3 = {};

  // ---- phase 1: K tiles from x_t (packed) ----
  for (int kt = 0; kt < 32; ++kt) {
    v16bf a = __builtin_bit_cast(v16bf, *(const us16*)(xfr + (size_t)kt * FRAG_H));
    const unsigned short* wp = wbase + (size_t)kt * FRAG_H;
    v16bf w0 = __builtin_bit_cast(v16bf, *(const us16*)(wp));
    v16bf w1 = __builtin_bit_cast(v16bf, *(const us16*)(wp + gstride));
    v16bf w2 = __builtin_bit_cast(v16bf, *(const us16*)(wp + 2 * gstride));
    v16bf w3 = __builtin_bit_cast(v16bf, *(const us16*)(wp + 3 * gstride));
    acc0 = __builtin_amdgcn_wmma_f32_16x16x32_bf16(false, a, false, w0, (short)0, acc0, false, false);
    acc1 = __builtin_amdgcn_wmma_f32_16x16x32_bf16(false, a, false, w1, (short)0, acc1, false, false);
    acc2 = __builtin_amdgcn_wmma_f32_16x16x32_bf16(false, a, false, w2, (short)0, acc2, false, false);
    acc3 = __builtin_amdgcn_wmma_f32_16x16x32_bf16(false, a, false, w3, (short)0, acc3, false, false);
  }
  // ---- phase 2: K tiles from h_prev (packed) ----
  for (int kt = 0; kt < 32; ++kt) {
    v16bf a = __builtin_bit_cast(v16bf, *(const us16*)(hfr + (size_t)kt * FRAG_H));
    const unsigned short* wp = wbase + (size_t)(32 + kt) * FRAG_H;
    v16bf w0 = __builtin_bit_cast(v16bf, *(const us16*)(wp));
    v16bf w1 = __builtin_bit_cast(v16bf, *(const us16*)(wp + gstride));
    v16bf w2 = __builtin_bit_cast(v16bf, *(const us16*)(wp + 2 * gstride));
    v16bf w3 = __builtin_bit_cast(v16bf, *(const us16*)(wp + 3 * gstride));
    acc0 = __builtin_amdgcn_wmma_f32_16x16x32_bf16(false, a, false, w0, (short)0, acc0, false, false);
    acc1 = __builtin_amdgcn_wmma_f32_16x16x32_bf16(false, a, false, w1, (short)0, acc1, false, false);
    acc2 = __builtin_amdgcn_wmma_f32_16x16x32_bf16(false, a, false, w2, (short)0, acc2, false, false);
    acc3 = __builtin_amdgcn_wmma_f32_16x16x32_bf16(false, a, false, w3, (short)0, acc3, false, false);
  }

  // ---- elementwise; C/D layout: lane->N, VGPR r -> M = r + 8*hsel ----
  const float Bi = bi[ncol], Bf = bfv[ncol], Bc = bc[ncol], Bo = bo[ncol];
  // packed-A destination coords for h_next element (mrow, ncol):
  const int kt_d = ncol >> 5;
  const int hs_d = (idx16 >> 3) & 1;                  // ((ncol&31)>>3)&1
  const int hh_d = (idx16 & 7) + 8 * (strip & 1);     // (ncol&7) + 8*((ncol>>4)&1)
  unsigned short* hdst = h_next + ((size_t)(mt * 32 + kt_d) * 32 + (hs_d * 16 + hsel * 8)) * 16 + hh_d;
#pragma unroll
  for (int r = 0; r < 8; ++r) {
    const int mrow = mt * 16 + hsel * 8 + r;
    const size_t o = (size_t)mrow * HID + ncol;
    float iv = sig_(acc0[r] + Bi);
    float fv = sig_(acc1[r] + Bf);
    float gv = tanh_(acc2[r] + Bc);
    float ov = sig_(acc3[r] + Bo);
    float cn = fv * cst[o] + iv * gv;
    float hn = ov * tanh_(cn);
    cst[o] = cn;
    hdst[(size_t)r * 16] = f2bf(hn);                  // packed-A layout, lane_dst stride 16 halves
    hseq[(size_t)t * BATCH * HID + o] = hn;
    if (last) { hT[o] = hn; cT[o] = cn; }
  }
}

// ---------------------------------------------------------------------------
// Fallback timestep (small workspace): convert x fp32->bf16 in-register.
// ---------------------------------------------------------------------------
__global__ void lstm_step_conv(const float* __restrict__ x,
                               const float* __restrict__ bi, const float* __restrict__ bfv,
                               const float* __restrict__ bc, const float* __restrict__ bo,
                               const unsigned short* __restrict__ Wb,
                               const unsigned short* __restrict__ h_prev,
                               unsigned short* __restrict__ h_next,
                               float* __restrict__ cst,
                               float* __restrict__ hseq,
                               float* __restrict__ hT, float* __restrict__ cT,
                               int t, int last) {
  const int lane  = threadIdx.x & 31;
  const int mt    = threadIdx.x >> 5;
  const int strip = blockIdx.x;
  const int idx16 = lane & 15;
  const int hsel  = lane >> 4;
  const int arow  = mt * 16 + idx16;
  const int ncol  = strip * 16 + idx16;
  const int kh    = hsel * 8;

  const float*          xrow = x + (size_t)t * BATCH * HID + (size_t)arow * HID;
  const unsigned short* hfr  = h_prev + ((size_t)mt * 32) * FRAG_H + (size_t)lane * 16;

  const size_t gstride = (size_t)NSTRIP * NKT * FRAG_H;
  const unsigned short* wbase = Wb + (size_t)strip * NKT * FRAG_H + (size_t)lane * 16;

  v8f acc0 = {}, acc1 = {}, acc2 = {}, acc3 = {};

  for (int kt = 0; kt < 32; ++kt) {
    const float* p = xrow + kt * 32 + kh;
    f4 f0 = *(const f4*)(p);
    f4 f1 = *(const f4*)(p + 4);
    f4 f2 = *(const f4*)(p + 16);
    f4 f3 = *(const f4*)(p + 20);
    us16 au;
#pragma unroll
    for (int j = 0; j < 4; ++j) {
      au[j] = f2bf(f0[j]); au[4 + j] = f2bf(f1[j]);
      au[8 + j] = f2bf(f2[j]); au[12 + j] = f2bf(f3[j]);
    }
    v16bf a = __builtin_bit_cast(v16bf, au);
    const unsigned short* wp = wbase + (size_t)kt * FRAG_H;
    v16bf w0 = __builtin_bit_cast(v16bf, *(const us16*)(wp));
    v16bf w1 = __builtin_bit_cast(v16bf, *(const us16*)(wp + gstride));
    v16bf w2 = __builtin_bit_cast(v16bf, *(const us16*)(wp + 2 * gstride));
    v16bf w3 = __builtin_bit_cast(v16bf, *(const us16*)(wp + 3 * gstride));
    acc0 = __builtin_amdgcn_wmma_f32_16x16x32_bf16(false, a, false, w0, (short)0, acc0, false, false);
    acc1 = __builtin_amdgcn_wmma_f32_16x16x32_bf16(false, a, false, w1, (short)0, acc1, false, false);
    acc2 = __builtin_amdgcn_wmma_f32_16x16x32_bf16(false, a, false, w2, (short)0, acc2, false, false);
    acc3 = __builtin_amdgcn_wmma_f32_16x16x32_bf16(false, a, false, w3, (short)0, acc3, false, false);
  }
  for (int kt = 0; kt < 32; ++kt) {
    v16bf a = __builtin_bit_cast(v16bf, *(const us16*)(hfr + (size_t)kt * FRAG_H));
    const unsigned short* wp = wbase + (size_t)(32 + kt) * FRAG_H;
    v16bf w0 = __builtin_bit_cast(v16bf, *(const us16*)(wp));
    v16bf w1 = __builtin_bit_cast(v16bf, *(const us16*)(wp + gstride));
    v16bf w2 = __builtin_bit_cast(v16bf, *(const us16*)(wp + 2 * gstride));
    v16bf w3 = __builtin_bit_cast(v16bf, *(const us16*)(wp + 3 * gstride));
    acc0 = __builtin_amdgcn_wmma_f32_16x16x32_bf16(false, a, false, w0, (short)0, acc0, false, false);
    acc1 = __builtin_amdgcn_wmma_f32_16x16x32_bf16(false, a, false, w1, (short)0, acc1, false, false);
    acc2 = __builtin_amdgcn_wmma_f32_16x16x32_bf16(false, a, false, w2, (short)0, acc2, false, false);
    acc3 = __builtin_amdgcn_wmma_f32_16x16x32_bf16(false, a, false, w3, (short)0, acc3, false, false);
  }

  const float Bi = bi[ncol], Bf = bfv[ncol], Bc = bc[ncol], Bo = bo[ncol];
  const int kt_d = ncol >> 5;
  const int hs_d = (idx16 >> 3) & 1;
  const int hh_d = (idx16 & 7) + 8 * (strip & 1);
  unsigned short* hdst = h_next + ((size_t)(mt * 32 + kt_d) * 32 + (hs_d * 16 + hsel * 8)) * 16 + hh_d;
#pragma unroll
  for (int r = 0; r < 8; ++r) {
    const int mrow = mt * 16 + hsel * 8 + r;
    const size_t o = (size_t)mrow * HID + ncol;
    float iv = sig_(acc0[r] + Bi);
    float fv = sig_(acc1[r] + Bf);
    float gv = tanh_(acc2[r] + Bc);
    float ov = sig_(acc3[r] + Bo);
    float cn = fv * cst[o] + iv * gv;
    float hn = ov * tanh_(cn);
    cst[o] = cn;
    hdst[(size_t)r * 16] = f2bf(hn);
    hseq[(size_t)t * BATCH * HID + o] = hn;
    if (last) { hT[o] = hn; cT[o] = cn; }
  }
}

extern "C" void kernel_launch(void* const* d_in, const int* in_sizes, int n_in,
                              void* d_out, int out_size, void* d_ws, size_t ws_size,
                              hipStream_t stream) {
  const float* x   = (const float*)d_in[0];
  const float* Wi  = (const float*)d_in[1];
  const float* bi  = (const float*)d_in[2];
  const float* Wf  = (const float*)d_in[3];
  const float* bf_ = (const float*)d_in[4];
  const float* Wc  = (const float*)d_in[5];
  const float* bc  = (const float*)d_in[6];
  const float* Wo  = (const float*)d_in[7];
  const float* bo  = (const float*)d_in[8];

  // ws layout: Wb 16MB | hb0 128KB | hb1 128KB | c 256KB | xb 64MB (optional)
  unsigned short* Wb  = (unsigned short*)d_ws;
  unsigned short* hb0 = Wb + (size_t)4 * 1024 * 2048;
  unsigned short* hb1 = hb0 + (size_t)BATCH * HID;
  float*          cst = (float*)(hb1 + (size_t)BATCH * HID);
  unsigned short* xb  = (unsigned short*)(cst + (size_t)BATCH * HID);
  const size_t need_full = ((char*)(xb + (size_t)T_STEPS * XSTEP)) - (char*)d_ws;
  const bool use_packed_x = ws_size >= need_full;

  float* out  = (float*)d_out;
  float* hseq = out;                                    // [T,B,H]
  float* hT   = out + (size_t)T_STEPS * BATCH * HID;    // [B,H]
  float* cT   = hT + (size_t)BATCH * HID;               // [B,H]

  pack_w<<<2048, 256, 0, stream>>>(Wi, Wf, Wc, Wo, Wb);
  init_state<<<256, 256, 0, stream>>>(hb0, cst);
  if (use_packed_x) {
    pack_x<<<8192, 256, 0, stream>>>(x, xb);
  }

  for (int t = 0; t < T_STEPS; ++t) {
    const unsigned short* hp = (t & 1) ? hb1 : hb0;
    unsigned short*       hn = (t & 1) ? hb0 : hb1;
    const int last = (t == T_STEPS - 1) ? 1 : 0;
    if (use_packed_x) {
      lstm_step<<<64, 128, 0, stream>>>(xb, bi, bf_, bc, bo, Wb, hp, hn, cst,
                                        hseq, hT, cT, t, last);
    } else {
      lstm_step_conv<<<64, 128, 0, stream>>>(x, bi, bf_, bc, bo, Wb, hp, hn, cst,
                                             hseq, hT, cT, t, last);
    }
  }
}
